// MNDRE_10960756539452
// MI455X (gfx1250) — matrix operand order repaired and checked by reading, not compiled, not run on hardware
//
#include <hip/hip_runtime.h>

#define PP 8
#define DD 64
#define NWIN 16
#define WW 16
#define NPAIR 28
#define LSIG 36
#define OUTD 32
#define TLEN 257

typedef __attribute__((ext_vector_type(16))) __bf16 v16bf;
typedef __attribute__((ext_vector_type(8)))  float  v8f;

union FragU {
    uint4 q[2];
    v16bf v;
};

__device__ __forceinline__ unsigned short f2bf(float f) {
    union { float f; unsigned u; } x; x.f = f;
    unsigned r = x.u + 0x7FFFu + ((x.u >> 16) & 1u);  // round-to-nearest-even
    return (unsigned short)(r >> 16);
}

__device__ __forceinline__ float lipsw(float x) {
    float s = 1.0f / (1.0f + __expf(-x));
    return 0.909f * x * s;
}
__device__ __forceinline__ float lipswd(float x) {
    float s = 1.0f / (1.0f + __expf(-x));
    return 0.909f * s * (1.0f + x * (1.0f - s));
}

// A fragment (16x32 bf16) from row-major u16 source, leading dim ld.
// Lane layout (ISA 7.12.2): m = lane&15, half = lane>>4;
// elems 0..7 <- K = 8*half..8*half+7 ; elems 8..15 <- K = 16+8*half..23+8*half
__device__ __forceinline__ v16bf loadA(const unsigned short* base, int ld,
                                       int row0, int k0, int lane) {
    int half = lane >> 4, m = lane & 15;
    const unsigned short* p = base + (row0 + m) * ld + k0;
    FragU f;
    f.q[0] = *(const uint4*)(p + 8 * half);
    f.q[1] = *(const uint4*)(p + 16 + 8 * half);
    return f.v;
}
// B fragment (32x16 bf16) from row-major [k][n] u16 source.
// Lane holds column n = n0 + (lane&15); elems 0..7 <- K = 8*half..+7, elems 8..15 <- K = 16+8*half..+7.
__device__ __forceinline__ v16bf loadB(const unsigned short* base, int ld,
                                       int k0, int n0, int lane) {
    int half = lane >> 4, nn = lane & 15;
    const unsigned short* p = base + k0 * ld + n0 + nn;
    FragU f;
    unsigned short t[16];
    #pragma unroll
    for (int s = 0; s < 8; ++s) {
        t[s]     = p[(8 * half + s) * ld];
        t[8 + s] = p[(16 + 8 * half + s) * ld];
    }
    __builtin_memcpy(&f, t, 32);
    return f.v;
}

__device__ __forceinline__ v8f wmma_bf16(v16bf a, v16bf b, v8f c) {
    return __builtin_amdgcn_wmma_f32_16x16x32_bf16(false, a, false, b,
                                                   (short)0, c, false, false);
}

// Pre-pass: convert vf_w2 (128x128) and vf_w3 (512x128) to bf16 in d_ws.
// Layout: w2b at [0,16384) elems, w3b at [16384, 81920).
__global__ void cvt_weights(const float* __restrict__ w2,
                            const float* __restrict__ w3,
                            unsigned short* __restrict__ ws) {
    int i = blockIdx.x * blockDim.x + threadIdx.x;
    if (i < 16384)       ws[i] = f2bf(w2[i]);
    else if (i < 81920)  ws[i] = f2bf(w3[i - 16384]);
}

__global__ __launch_bounds__(256) void cde_solve(
    const float* __restrict__ cv,
    const float* __restrict__ ic_w1, const float* __restrict__ ic_b1,
    const float* __restrict__ ic_w2, const float* __restrict__ ic_b2,
    const float* __restrict__ ic_w3, const float* __restrict__ ic_b3,
    const float* __restrict__ vf_w1, const float* __restrict__ vf_b1,
    const float* __restrict__ vf_w2, const float* __restrict__ vf_b2,
    const float* __restrict__ vf_w3, const float* __restrict__ vf_b3,
    const float* __restrict__ ro_w,  const float* __restrict__ ro_b,
    const unsigned short* __restrict__ wsb,
    float* __restrict__ out)
{
    __shared__ __align__(16) float xs[TLEN * PP];          // path
    __shared__ float logs[NWIN * LSIG];                    // log-signatures
    __shared__ float hid[(NWIN + 1) * DD];                 // hidden states
    __shared__ float ybase[DD], yc[DD], acc_s[DD];
    __shared__ float h1[128], d1[128], h2[128], d2[128];
    __shared__ float Fv[512], d3[512];
    __shared__ __align__(16) unsigned short Feb[64 * 16];  // F^T (padded to 16 cols), bf16
    __shared__ __align__(16) unsigned short Ubuf[128 * 64];// U1 then U2, bf16
    __shared__ __align__(16) unsigned short Jl[512 * 64];  // J = d3 .* T3, bf16
    __shared__ float Ml[64 * 64];                          // M[a*8+b][d]

    const int tid = threadIdx.x;
    const int lane = tid & 31;
    const int wave = tid >> 5;
    const int b = blockIdx.x;

    const unsigned short* w2b = wsb;
    const unsigned short* w3b = wsb + 16384;

    // ---- load path, zero Feb pad rows ----
    for (int i = tid; i < TLEN * PP; i += 256) xs[i] = cv[b * TLEN * PP + i];
    for (int i = tid; i < 64 * 16; i += 256) if ((i & 15) >= PP) Feb[i] = 0;
    __syncthreads();

    // ---- windowed depth-2 Lyndon log-signatures (16 threads, one per window) ----
    if (tid < NWIN) {
        float pre[PP], lev1[PP], accp[NPAIR];
        #pragma unroll
        for (int i = 0; i < PP; ++i) { pre[i] = 0.f; lev1[i] = 0.f; }
        #pragma unroll
        for (int p = 0; p < NPAIR; ++p) accp[p] = 0.f;
        const float* xw = xs + tid * WW * PP;
        for (int k = 0; k < WW; ++k) {
            float dxk[PP];
            #pragma unroll
            for (int i = 0; i < PP; ++i) dxk[i] = xw[(k + 1) * PP + i] - xw[k * PP + i];
            int p = 0;
            #pragma unroll
            for (int ii = 0; ii < PP; ++ii)
                #pragma unroll
                for (int jj = ii + 1; jj < PP; ++jj, ++p)
                    accp[p] += pre[ii] * dxk[jj] - pre[jj] * dxk[ii];
            #pragma unroll
            for (int i = 0; i < PP; ++i) { pre[i] += dxk[i]; lev1[i] += dxk[i]; }
        }
        #pragma unroll
        for (int i = 0; i < PP; ++i) logs[tid * LSIG + i] = lev1[i];
        #pragma unroll
        for (int p = 0; p < NPAIR; ++p) logs[tid * LSIG + PP + p] = 0.5f * accp[p];
    }
    __syncthreads();

    // ---- ic_mlp: y0 = W3 ls(W2 ls(W1 x0 + b1) + b2) + b3 ----
    if (tid < 128) {
        float s = ic_b1[tid];
        for (int j = 0; j < PP; ++j) s += ic_w1[tid * PP + j] * xs[j];
        h1[tid] = lipsw(s);
    }
    __syncthreads();
    if (tid < 128) {
        float s = ic_b2[tid];
        for (int j = 0; j < 128; ++j) s += ic_w2[tid * 128 + j] * h1[j];
        h2[tid] = lipsw(s);
    }
    __syncthreads();
    if (tid < DD) {
        float s = ic_b3[tid];
        for (int j = 0; j < 128; ++j) s += ic_w3[tid * 128 + j] * h2[j];
        ybase[tid] = s; yc[tid] = s; hid[tid] = s;
    }
    __syncthreads();

    // ---- RK4 scan over 16 windows ----
    for (int step = 0; step < NWIN; ++step) {
        const float* v = logs + step * LSIG;
        for (int st = 0; st < 4; ++st) {
            // (a) z1 = W1 yc + b1  -> h1, d1
            if (tid < 128) {
                float s = vf_b1[tid];
                const float* wr = vf_w1 + tid * DD;
                for (int j = 0; j < DD; ++j) s += wr[j] * yc[j];
                h1[tid] = lipsw(s); d1[tid] = lipswd(s);
            }
            __syncthreads();
            // (b) z2 -> h2, d2 ; build U1[i][j] = d1[i]*W1[i][j] (bf16, [k][n] layout)
            if (tid < 128) {
                float s = vf_b2[tid];
                const float* wr = vf_w2 + tid * 128;
                for (int j = 0; j < 128; ++j) s += wr[j] * h1[j];
                h2[tid] = lipsw(s); d2[tid] = lipswd(s);
            }
            for (int idx = tid; idx < 128 * DD; idx += 256)
                Ubuf[idx] = f2bf(d1[idx >> 6] * vf_w1[idx]);
            __syncthreads();
            // (c) z3 -> F = tanh(z3), d3 = 1 - F^2, Feb = F^T bf16
            for (int o = tid; o < 512; o += 256) {
                float s = vf_b3[o];
                const float* wr = vf_w3 + o * 128;
                for (int j = 0; j < 128; ++j) s += wr[j] * h2[j];
                float t = tanhf(s);
                Fv[o] = t; d3[o] = 1.0f - t * t;
                Feb[(o & 63) * 16 + (o >> 6)] = f2bf(t);
            }
            __syncthreads();
            // (d) T2 = W2 @ U1 via WMMA (8 waves x 4 N-tiles), then U2 = d2 .* T2
            {
                int mt = wave;
                v8f a0 = {}, a1 = {}, a2 = {}, a3 = {};
                #pragma unroll
                for (int kk = 0; kk < 4; ++kk) {
                    v16bf af = loadA(w2b, 128, mt * 16, kk * 32, lane);
                    a0 = wmma_bf16(af, loadB(Ubuf, 64, kk * 32, 0,  lane), a0);
                    a1 = wmma_bf16(af, loadB(Ubuf, 64, kk * 32, 16, lane), a1);
                    a2 = wmma_bf16(af, loadB(Ubuf, 64, kk * 32, 32, lane), a2);
                    a3 = wmma_bf16(af, loadB(Ubuf, 64, kk * 32, 48, lane), a3);
                }
                __syncthreads();  // all U1 reads done before Ubuf is overwritten
                int half = lane >> 4, n = lane & 15;
                #pragma unroll
                for (int r = 0; r < 8; ++r) {
                    int i2 = mt * 16 + r + 8 * half;
                    float sc = d2[i2];
                    Ubuf[i2 * 64 +  0 + n] = f2bf(sc * a0[r]);
                    Ubuf[i2 * 64 + 16 + n] = f2bf(sc * a1[r]);
                    Ubuf[i2 * 64 + 32 + n] = f2bf(sc * a2[r]);
                    Ubuf[i2 * 64 + 48 + n] = f2bf(sc * a3[r]);
                }
            }
            __syncthreads();
            // (e) T3 = W3 @ U2 via WMMA (8 waves x 4 M-rows x 4 N-tiles); J = d3 .* T3
            for (int q = 0; q < 4; ++q) {
                int mt = wave * 4 + q;
                v8f a0 = {}, a1 = {}, a2 = {}, a3 = {};
                #pragma unroll
                for (int kk = 0; kk < 4; ++kk) {
                    v16bf af = loadA(w3b, 128, mt * 16, kk * 32, lane);
                    a0 = wmma_bf16(af, loadB(Ubuf, 64, kk * 32, 0,  lane), a0);
                    a1 = wmma_bf16(af, loadB(Ubuf, 64, kk * 32, 16, lane), a1);
                    a2 = wmma_bf16(af, loadB(Ubuf, 64, kk * 32, 32, lane), a2);
                    a3 = wmma_bf16(af, loadB(Ubuf, 64, kk * 32, 48, lane), a3);
                }
                int half = lane >> 4, n = lane & 15;
                #pragma unroll
                for (int r = 0; r < 8; ++r) {
                    int o = mt * 16 + r + 8 * half;
                    float sc = d3[o];
                    Jl[o * 64 +  0 + n] = f2bf(sc * a0[r]);
                    Jl[o * 64 + 16 + n] = f2bf(sc * a1[r]);
                    Jl[o * 64 + 32 + n] = f2bf(sc * a2[r]);
                    Jl[o * 64 + 48 + n] = f2bf(sc * a3[r]);
                }
            }
            __syncthreads();
            // (f) M_a[d][b] = sum_e J[a*64+d][e] * F[b][e]  (A = Jl, B = Feb)
            #pragma unroll
            for (int i = 0; i < 4; ++i) {
                int job = wave * 4 + i;
                int a = job >> 2, dt = job & 3;
                v8f mc = {};
                #pragma unroll
                for (int kk = 0; kk < 2; ++kk) {
                    v16bf af = loadA(Jl, 64, a * 64 + dt * 16, kk * 32, lane);
                    v16bf bf = loadB(Feb, 16, kk * 32, 0, lane);
                    mc = wmma_bf16(af, bf, mc);
                }
                int half = lane >> 4, bb = lane & 15;
                if (bb < PP) {
                    #pragma unroll
                    for (int r = 0; r < 8; ++r) {
                        int d = dt * 16 + r + 8 * half;
                        Ml[(a * PP + bb) * 64 + d] = mc[r];
                    }
                }
            }
            __syncthreads();
            // (g) g = [F ; lev2]^T @ v, RK4 stage update
            if (tid < DD) {
                float g = 0.0f;
                #pragma unroll
                for (int a = 0; a < PP; ++a) g += Fv[a * 64 + tid] * v[a];
                int p = 0;
                #pragma unroll
                for (int ii = 0; ii < PP; ++ii)
                    #pragma unroll
                    for (int jj = ii + 1; jj < PP; ++jj, ++p)
                        g += v[PP + p] *
                             (Ml[(jj * PP + ii) * 64 + tid] - Ml[(ii * PP + jj) * 64 + tid]);
                float yb = ybase[tid];
                if (st == 0)      { acc_s[tid] = g;          yc[tid] = yb + 0.5f * g; }
                else if (st == 1) { acc_s[tid] += 2.0f * g;  yc[tid] = yb + 0.5f * g; }
                else if (st == 2) { acc_s[tid] += 2.0f * g;  yc[tid] = yb + g; }
                else {
                    float yn = yb + (acc_s[tid] + g) * (1.0f / 6.0f);
                    ybase[tid] = yn; yc[tid] = yn;
                    hid[(step + 1) * DD + tid] = yn;
                }
            }
            __syncthreads();
        }
    }

    // ---- readout: out[b][t][o] = hid[t] . ro_w[o] + ro_b[o] ----
    for (int idx = tid; idx < (NWIN + 1) * OUTD; idx += 256) {
        int t = idx >> 5, o = idx & 31;
        float s = ro_b[o];
        for (int dd2 = 0; dd2 < DD; ++dd2) s += hid[t * DD + dd2] * ro_w[o * DD + dd2];
        out[b * (NWIN + 1) * OUTD + idx] = s;
    }
}

extern "C" void kernel_launch(void* const* d_in, const int* in_sizes, int n_in,
                              void* d_out, int out_size, void* d_ws, size_t ws_size,
                              hipStream_t stream) {
    const float* cv    = (const float*)d_in[0];
    const float* ic_w1 = (const float*)d_in[1];
    const float* ic_b1 = (const float*)d_in[2];
    const float* ic_w2 = (const float*)d_in[3];
    const float* ic_b2 = (const float*)d_in[4];
    const float* ic_w3 = (const float*)d_in[5];
    const float* ic_b3 = (const float*)d_in[6];
    const float* vf_w1 = (const float*)d_in[7];
    const float* vf_b1 = (const float*)d_in[8];
    const float* vf_w2 = (const float*)d_in[9];
    const float* vf_b2 = (const float*)d_in[10];
    const float* vf_w3 = (const float*)d_in[11];
    const float* vf_b3 = (const float*)d_in[12];
    const float* ro_w  = (const float*)d_in[13];
    const float* ro_b  = (const float*)d_in[14];
    float* out = (float*)d_out;
    unsigned short* wsb = (unsigned short*)d_ws;

    int B = in_sizes[0] / (TLEN * PP);

    cvt_weights<<<(81920 + 255) / 256, 256, 0, stream>>>(vf_w2, vf_w3, wsb);
    cde_solve<<<B, 256, 0, stream>>>(cv, ic_w1, ic_b1, ic_w2, ic_b2, ic_w3, ic_b3,
                                     vf_w1, vf_b1, vf_w2, vf_b2, vf_w3, vf_b3,
                                     ro_w, ro_b, wsb, out);
}
